// LiveSR_57947698758017
// MI455X (gfx1250) — matrix operand order repaired and checked by reading, not compiled, not run on hardware
//
#include <hip/hip_runtime.h>
#include <hip/hip_bf16.h>

// ---------------------------------------------------------------------------
// LiveSR expert-routed SR on MI455X (gfx1250, wave32).
// All convs = im2col GEMM on v_wmma_f32_16x16x32_f16 with:
//   * weights pre-padded to f16 [Mpad][Kpad] (branch-free A: 2x b128/lane)
//   * activations in zero-halo f16 [40][H+2][W+2] global layout
//   * per-block LDS staging via global_load_async_to_lds_b64 (ASYNCcnt path)
//   * branch-free ds gathers for the im2col B operand
// ---------------------------------------------------------------------------

typedef __attribute__((ext_vector_type(16))) _Float16 v16h;
typedef __attribute__((ext_vector_type(8)))  _Float16 v8h;
typedef __attribute__((ext_vector_type(8)))  float    v8f;
typedef __attribute__((ext_vector_type(16))) int      v16i;

#define NSAMP   128
#define NEXPERT 10
#define NBLK    8
#define FCH     36
#define FEATD   512
#define CPAD    40      // channel pad: k<352 -> ci<40, zeroed beyond 36
#define KPAD    352     // 324 -> 11 k-steps of 32

// ---------------- Stage 0: classifier feature (3->512 conv, relu, mean) ----
__global__ void __launch_bounds__(256)
k_classifier(const float* __restrict__ inputs, const float* __restrict__ cls_w,
             const float* __restrict__ cls_b, float* __restrict__ feat) {
  __shared__ float s_in[3 * 32 * 32];
  const int s = blockIdx.x;
  for (int i = threadIdx.x; i < 3 * 32 * 32; i += blockDim.x)
    s_in[i] = inputs[s * 3 * 32 * 32 + i];
  __syncthreads();

  for (int c = threadIdx.x; c < FEATD; c += blockDim.x) {
    float wl[27];
#pragma unroll
    for (int i = 0; i < 27; ++i) wl[i] = cls_w[c * 27 + i];
    const float bb = cls_b[c];
    float sum = 0.0f;
    for (int y = 0; y < 32; ++y) {
      for (int x = 0; x < 32; ++x) {
        float a = bb;
#pragma unroll
        for (int ci = 0; ci < 3; ++ci)
#pragma unroll
          for (int ky = 0; ky < 3; ++ky) {
            const int yy = y + ky - 1;
            if (yy < 0 || yy > 31) continue;
#pragma unroll
            for (int kx = 0; kx < 3; ++kx) {
              const int xx = x + kx - 1;
              if (xx < 0 || xx > 31) continue;
              a += wl[ci * 9 + ky * 3 + kx] * s_in[(ci * 32 + yy) * 32 + xx];
            }
          }
        sum += (a > 0.0f) ? a : 0.0f;
      }
    }
    feat[s * FEATD + c] = sum * (1.0f / 1024.0f);
  }
}

// ---------------- Stage 1: k-means top-1 routing ---------------------------
__global__ void k_route(const float* __restrict__ feat,
                        const float* __restrict__ centers,
                        int* __restrict__ assign) {
  const int n = threadIdx.x;
  if (n >= NSAMP) return;
  const float* f = feat + n * FEATD;
  int best = 0;
  float bestd = 3.4e38f;
  for (int k = 0; k < NEXPERT; ++k) {
    const float* c = centers + k * FEATD;
    float d = 0.0f;
    for (int i = 0; i < FEATD; ++i) {
      const float t = f[i] - c[i];
      d += t * t;
    }
    if (d < bestd) { bestd = d; best = k; }  // first-min == argmax(1/d)
  }
  assign[n] = best;
}

// ---------------- Weight pre-pad: f32 [c][M][Keff] -> f16 [c][Mpad][Kpad] --
__global__ void k_prepw(const float* __restrict__ src,
                        _Float16* __restrict__ dst, int M, int Keff, int Mpad,
                        int Kpad, int nconv) {
  const int total = nconv * Mpad * Kpad;
  for (int i = blockIdx.x * blockDim.x + threadIdx.x; i < total;
       i += gridDim.x * blockDim.x) {
    const int c = i / (Mpad * Kpad);
    const int rem = i - c * (Mpad * Kpad);
    const int m = rem / Kpad, k = rem - m * Kpad;
    float v = 0.0f;
    if (m < M && k < Keff) v = src[((size_t)c * M + m) * Keff + k];
    dst[i] = (_Float16)v;
  }
}

// ---------------- fragment helpers -----------------------------------------
// A (16x32): lane row m = m0 + (lane&15); elems j0..7 = w[kk+hi*8 .. +7],
//            j8..15 = w[kk+16+hi*8 .. +7]  -> two aligned b128 loads.
__device__ __forceinline__ v16h load_A(const _Float16* __restrict__ wrow,
                                       int kk, int hi) {
  const _Float16* p0 = wrow + kk + hi * 8;
  const v8h a0 = *(const v8h*)p0;
  const v8h a1 = *(const v8h*)(p0 + 16);
  return __builtin_shufflevector(a0, a1, 0, 1, 2, 3, 4, 5, 6, 7, 8, 9, 10, 11,
                                 12, 13, 14, 15);
}

// ---------------- generic conv kernel --------------------------------------
// EPI 0: store halo f16 (mode 1 = relu, mode 2 = += aux at same idx)
// EPI 1: pixel-shuffle store into 2x halo f16
// EPI 2: f32 store to final output (M=3)
template <int WH, int RG, int EPI>
__global__ void __launch_bounds__(256)
k_conv(const _Float16* __restrict__ wp, const float* __restrict__ bias_base,
       const int* __restrict__ assign, const _Float16* __restrict__ src,
       const _Float16* __restrict__ aux, _Float16* __restrict__ dst16,
       float* __restrict__ dst32, int Mreal, int Mpad, int Kpad, int wconvs,
       int widx, int mblocks, int mode) {
  constexpr int W = WH - 2;
  constexpr int LOGW = (WH == 34) ? 5 : ((WH == 66) ? 6 : 7);
  constexpr int RO = RG - 2;            // output rows per block
  constexpr int MT = (EPI == 2) ? 1 : 3;
  constexpr int NT = (RO * W) / 16;     // n-tiles per block
  constexpr int JPW = NT / 8;           // jobs per wave (8 waves/block)

  __shared__ alignas(16) _Float16 s_act[CPAD * RG * WH];
  __shared__ alignas(64) int s_tbl[KPAD];

  const int s = blockIdx.y;
  const int e = assign[s];
  const int rg = blockIdx.x / mblocks;
  const int mb = blockIdx.x - rg * mblocks;
  const int m0base = mb * 16 * MT;

  const _Float16* wconv =
      wp + ((size_t)e * wconvs + widx) * (size_t)Mpad * Kpad;
  const float* bias = bias_base + ((size_t)e * wconvs + widx) * Mreal;

  // k -> LDS offset table (relative to pixel base)
  for (int k = threadIdx.x; k < Kpad; k += blockDim.x) {
    const int ci = k / 9, r = k - ci * 9, q = r / 3;
    s_tbl[k] = (ci * RG + (q - 1)) * WH + (r - q * 3) - 1;
  }

  // stage halo rows [rg*RO, rg*RO+RG) of all 40 channels via async DMA:
  // per-channel slab is RG*WH halfs, contiguous in global and LDS, and both
  // addresses are 8B-aligned => b64 chunks (ASYNCcnt-tracked, no VGPR data).
  constexpr int CH_HALFS = RG * WH;
  constexpr int CH_CHUNKS = CH_HALFS / 4;       // b64 chunks per channel
  constexpr int CHUNKS = CPAD * CH_CHUNKS;
  const _Float16* srcs =
      src + (size_t)s * CPAD * WH * WH + (size_t)rg * RO * WH;
  for (int c = threadIdx.x; c < CHUNKS; c += blockDim.x) {
    const int ci = c / CH_CHUNKS;
    const int rem = c - ci * CH_CHUNKS;
    const _Float16* g = srcs + (size_t)ci * WH * WH + (size_t)rem * 4;
    const unsigned lds = (unsigned)(uintptr_t)(&s_act[c * 4]);
    asm volatile("global_load_async_to_lds_b64 %0, %1, off"
                 :: "v"(lds), "v"(g)
                 : "memory");
  }
  asm volatile("s_wait_asynccnt 0" ::: "memory");
  __syncthreads();

  const int lane = threadIdx.x & 31;
  const int l = lane & 15, hi = lane >> 4;
  const int wave = threadIdx.x >> 5;

  int pixbase[JPW], gy[JPW], gx[JPW];
#pragma unroll
  for (int u = 0; u < JPW; ++u) {
    const int p = (wave + u * 8) * 16 + l;
    const int ylr = p >> LOGW;
    const int x = p & (W - 1);
    pixbase[u] = (ylr + 1) * WH + x + 1;
    gy[u] = rg * RO + ylr;
    gx[u] = x;
  }

  v8f acc[JPW][MT];
#pragma unroll
  for (int u = 0; u < JPW; ++u)
#pragma unroll
    for (int t = 0; t < MT; ++t) acc[u][t] = (v8f){};

  for (int kk = 0; kk < Kpad; kk += 32) {
    v16h af[MT];
#pragma unroll
    for (int t = 0; t < MT; ++t)
      af[t] = load_A(wconv + (size_t)(m0base + 16 * t + l) * Kpad, kk, hi);

    typedef __attribute__((ext_vector_type(8))) int v8i_t;
    const v8i_t t0 = *(const v8i_t*)(s_tbl + kk + (hi << 3));
    const v8i_t t1 = *(const v8i_t*)(s_tbl + kk + 16 + (hi << 3));
    const v16i tv = __builtin_shufflevector(t0, t1, 0, 1, 2, 3, 4, 5, 6, 7, 8,
                                            9, 10, 11, 12, 13, 14, 15);
#pragma unroll
    for (int u = 0; u < JPW; ++u) {
      v16h bf;
#pragma unroll
      for (int j = 0; j < 16; ++j) bf[j] = s_act[tv[j] + pixbase[u]];
#pragma unroll
      for (int t = 0; t < MT; ++t)
        acc[u][t] = __builtin_amdgcn_wmma_f32_16x16x32_f16(
            false, af[t], false, bf, (short)0, acc[u][t], false, false);
    }
  }

#pragma unroll
  for (int u = 0; u < JPW; ++u) {
    if constexpr (EPI == 0) {
      _Float16* d = dst16 + (size_t)s * CPAD * WH * WH;
#pragma unroll
      for (int t = 0; t < MT; ++t)
#pragma unroll
        for (int v = 0; v < 8; ++v) {
          const int m = m0base + 16 * t + v + hi * 8;
          if (m < Mreal) {
            const size_t di = ((size_t)m * WH + gy[u] + 1) * WH + gx[u] + 1;
            float val = acc[u][t][v] + bias[m];
            if (mode == 1) {
              val = (val > 0.0f) ? val : 0.0f;
            } else if (mode == 2) {
              const _Float16* a = aux + (size_t)s * CPAD * WH * WH;
              val += (float)a[di];
            }
            d[di] = (_Float16)val;
          }
        }
    } else if constexpr (EPI == 1) {
      constexpr int WH2 = 2 * WH - 2;
      _Float16* d = dst16 + (size_t)s * CPAD * WH2 * WH2;
#pragma unroll
      for (int t = 0; t < MT; ++t)
#pragma unroll
        for (int v = 0; v < 8; ++v) {
          const int m = m0base + 16 * t + v + hi * 8;  // < 144 always
          const float val = acc[u][t][v] + bias[m];
          const int c2 = m >> 2, ry = (m >> 1) & 1, rx = m & 1;
          const int Y = (gy[u] << 1) | ry, X = (gx[u] << 1) | rx;
          d[((size_t)c2 * WH2 + Y + 1) * WH2 + X + 1] = (_Float16)val;
        }
    } else {
      float* d = dst32 + (size_t)s * 3 * W * W;
#pragma unroll
      for (int v = 0; v < 8; ++v) {
        const int m = v + hi * 8;
        if (m < 3)
          d[(size_t)m * W * W + gy[u] * W + gx[u]] = acc[u][0][v] + bias[m];
      }
    }
  }
}

// ---------------- head conv 3->36 (f32 src staged to zero-padded LDS) ------
__global__ void __launch_bounds__(256)
k_head(const float* __restrict__ inputs, const _Float16* __restrict__ wp,
       const float* __restrict__ head_b, const int* __restrict__ assign,
       _Float16* __restrict__ h16, _Float16* __restrict__ b16) {
  constexpr int WH = 34, RG = 10, RO = 8, KP = 32;
  __shared__ _Float16 s_act[4 * RG * WH];
  __shared__ alignas(64) int s_tbl[KP];

  const int s = blockIdx.y;
  const int e = assign[s];
  const int rg = blockIdx.x;

  const _Float16* wconv = wp + (size_t)e * 48 * KP;
  const float* bias = head_b + e * FCH;

  for (int k = threadIdx.x; k < KP; k += blockDim.x) {
    const int ci = k / 9, r = k - ci * 9, q = r / 3;
    s_tbl[k] = (ci * RG + (q - 1)) * WH + (r - q * 3) - 1;
  }
  for (int i = threadIdx.x; i < 4 * RG * WH; i += blockDim.x)
    s_act[i] = (_Float16)0.0f;
  __syncthreads();
  const float* in_s = inputs + (size_t)s * 3 * 1024;
  for (int i = threadIdx.x; i < 3 * RG * 32; i += blockDim.x) {
    const int ci = i / (RG * 32);
    const int rr = (i / 32) % RG;
    const int x = i & 31;
    const int gy0 = rg * RO + rr - 1;
    if (gy0 >= 0 && gy0 < 32)
      s_act[(ci * RG + rr) * WH + x + 1] =
          (_Float16)in_s[(ci * 32 + gy0) * 32 + x];
  }
  __syncthreads();

  const int lane = threadIdx.x & 31;
  const int l = lane & 15, hi = lane >> 4;
  const int wave = threadIdx.x >> 5;

  int pixbase[2], gy[2], gx[2];
#pragma unroll
  for (int u = 0; u < 2; ++u) {
    const int p = (wave + u * 8) * 16 + l;
    const int ylr = p >> 5, x = p & 31;
    pixbase[u] = (ylr + 1) * WH + x + 1;
    gy[u] = rg * RO + ylr;
    gx[u] = x;
  }

  v8f acc[2][3];
#pragma unroll
  for (int u = 0; u < 2; ++u)
#pragma unroll
    for (int t = 0; t < 3; ++t) acc[u][t] = (v8f){};

  {
    v16h af[3];
#pragma unroll
    for (int t = 0; t < 3; ++t)
      af[t] = load_A(wconv + (size_t)(16 * t + l) * KP, 0, hi);
    typedef __attribute__((ext_vector_type(8))) int v8i_t;
    const v8i_t t0 = *(const v8i_t*)(s_tbl + (hi << 3));
    const v8i_t t1 = *(const v8i_t*)(s_tbl + 16 + (hi << 3));
    const v16i tv = __builtin_shufflevector(t0, t1, 0, 1, 2, 3, 4, 5, 6, 7, 8,
                                            9, 10, 11, 12, 13, 14, 15);
#pragma unroll
    for (int u = 0; u < 2; ++u) {
      v16h bf;
#pragma unroll
      for (int j = 0; j < 16; ++j) bf[j] = s_act[tv[j] + pixbase[u]];
#pragma unroll
      for (int t = 0; t < 3; ++t)
        acc[u][t] = __builtin_amdgcn_wmma_f32_16x16x32_f16(
            false, af[t], false, bf, (short)0, acc[u][t], false, false);
    }
  }

#pragma unroll
  for (int u = 0; u < 2; ++u)
#pragma unroll
    for (int t = 0; t < 3; ++t)
#pragma unroll
      for (int v = 0; v < 8; ++v) {
        const int m = 16 * t + v + hi * 8;
        if (m < FCH) {
          const float val = acc[u][t][v] + bias[m];
          const size_t di =
              (size_t)s * CPAD * WH * WH + ((size_t)m * WH + gy[u] + 1) * WH +
              gx[u] + 1;
          h16[di] = (_Float16)val;
          b16[di] = (_Float16)val;
        }
      }
}

// ---------------------------------------------------------------------------
extern "C" void kernel_launch(void* const* d_in, const int* in_sizes, int n_in,
                              void* d_out, int out_size, void* d_ws,
                              size_t ws_size, hipStream_t stream) {
  (void)in_sizes; (void)n_in; (void)out_size; (void)ws_size;
  const float* inputs  = (const float*)d_in[0];
  const float* cls_w   = (const float*)d_in[1];
  const float* cls_b   = (const float*)d_in[2];
  const float* centers = (const float*)d_in[3];
  const float* head_w  = (const float*)d_in[4];
  const float* head_b  = (const float*)d_in[5];
  const float* bw1     = (const float*)d_in[6];
  const float* bb1     = (const float*)d_in[7];
  const float* bw2     = (const float*)d_in[8];
  const float* bb2     = (const float*)d_in[9];
  const float* body_w  = (const float*)d_in[10];
  const float* body_b  = (const float*)d_in[11];
  const float* up1_w   = (const float*)d_in[12];
  const float* up1_b   = (const float*)d_in[13];
  const float* up2_w   = (const float*)d_in[14];
  const float* up2_b   = (const float*)d_in[15];
  const float* tail_w  = (const float*)d_in[16];
  const float* tail_b  = (const float*)d_in[17];
  float* out = (float*)d_out;

  char* ws = (char*)d_ws;
  size_t off = 0;
  auto alloc = [&](size_t bytes) -> void* {
    off = (off + 255) & ~(size_t)255;
    void* p = ws + off;
    off += bytes;
    return p;
  };
  int*   assign = (int*)alloc(NSAMP * sizeof(int));
  float* feat   = (float*)alloc((size_t)NSAMP * FEATD * sizeof(float));
  // pre-padded f16 weights
  _Float16* wp_head = (_Float16*)alloc((size_t)NEXPERT * 48 * 32 * 2);
  _Float16* wp_bw1  = (_Float16*)alloc((size_t)NEXPERT * NBLK * 48 * KPAD * 2);
  _Float16* wp_bw2  = (_Float16*)alloc((size_t)NEXPERT * NBLK * 48 * KPAD * 2);
  _Float16* wp_body = (_Float16*)alloc((size_t)NEXPERT * 48 * KPAD * 2);
  _Float16* wp_up1  = (_Float16*)alloc((size_t)NEXPERT * 144 * KPAD * 2);
  _Float16* wp_up2  = (_Float16*)alloc((size_t)NEXPERT * 144 * KPAD * 2);
  _Float16* wp_tail = (_Float16*)alloc((size_t)NEXPERT * 16 * KPAD * 2);
  // zero-halo activations [40][H+2][W+2]
  const size_t sz34  = (size_t)NSAMP * CPAD * 34 * 34 * 2;
  const size_t sz66  = (size_t)NSAMP * CPAD * 66 * 66 * 2;
  const size_t sz130 = (size_t)NSAMP * CPAD * 130 * 130 * 2;
  _Float16* h16 = (_Float16*)alloc(sz34);
  _Float16* b16 = (_Float16*)alloc(sz34);
  _Float16* t16 = (_Float16*)alloc(sz34);
  _Float16* u1  = (_Float16*)alloc(sz66);
  _Float16* u2  = (_Float16*)alloc(sz130);

  // zero halos/pad-channels (graph-capturable memset nodes)
  hipMemsetAsync(h16, 0, sz34, stream);
  hipMemsetAsync(b16, 0, sz34, stream);
  hipMemsetAsync(t16, 0, sz34, stream);
  hipMemsetAsync(u1, 0, sz66, stream);
  hipMemsetAsync(u2, 0, sz130, stream);

  // weight pre-pad
  auto prep = [&](const float* s, _Float16* d, int M, int Ke, int Mp, int Kp,
                  int nc) {
    const int total = nc * Mp * Kp;
    int blocks = (total + 255) / 256;
    if (blocks > 2048) blocks = 2048;
    k_prepw<<<blocks, 256, 0, stream>>>(s, d, M, Ke, Mp, Kp, nc);
  };
  prep(head_w, wp_head, FCH, 27, 48, 32, NEXPERT);
  prep(bw1, wp_bw1, FCH, 324, 48, KPAD, NEXPERT * NBLK);
  prep(bw2, wp_bw2, FCH, 324, 48, KPAD, NEXPERT * NBLK);
  prep(body_w, wp_body, FCH, 324, 48, KPAD, NEXPERT);
  prep(up1_w, wp_up1, 144, 324, 144, KPAD, NEXPERT);
  prep(up2_w, wp_up2, 144, 324, 144, KPAD, NEXPERT);
  prep(tail_w, wp_tail, 3, 324, 16, KPAD, NEXPERT);

  // classifier + routing
  k_classifier<<<NSAMP, 256, 0, stream>>>(inputs, cls_w, cls_b, feat);
  k_route<<<1, 128, 0, stream>>>(feat, centers, assign);

  // head: 4 rowgroups per sample
  k_head<<<dim3(4, NSAMP), 256, 0, stream>>>(inputs, wp_head, head_b, assign,
                                             h16, b16);

  // 8 residual blocks: conv1(relu) b->t ; conv2(+res) t->b
  for (int i = 0; i < NBLK; ++i) {
    k_conv<34, 10, 0><<<dim3(4, NSAMP), 256, 0, stream>>>(
        wp_bw1, bb1, assign, b16, nullptr, t16, nullptr, FCH, 48, KPAD, NBLK,
        i, 1, 1);
    k_conv<34, 10, 0><<<dim3(4, NSAMP), 256, 0, stream>>>(
        wp_bw2, bb2, assign, t16, b16, b16, nullptr, FCH, 48, KPAD, NBLK, i,
        1, 2);
  }
  // body + global skip: b -> t (t = conv(b) + h)
  k_conv<34, 10, 0><<<dim3(4, NSAMP), 256, 0, stream>>>(
      wp_body, body_b, assign, b16, h16, t16, nullptr, FCH, 48, KPAD, 1, 0, 1,
      2);

  // up1 @32x32 (4 rowgroups x 3 mblocks) -> u1 [40][66][66]
  k_conv<34, 10, 1><<<dim3(12, NSAMP), 256, 0, stream>>>(
      wp_up1, up1_b, assign, t16, nullptr, u1, nullptr, 144, 144, KPAD, 1, 0,
      3, 0);
  // up2 @64x64 (8 rowgroups x 3 mblocks) -> u2 [40][130][130]
  k_conv<66, 10, 1><<<dim3(24, NSAMP), 256, 0, stream>>>(
      wp_up2, up2_b, assign, u1, nullptr, u2, nullptr, 144, 144, KPAD, 1, 0,
      3, 0);
  // tail @128x128 (32 rowgroups) -> f32 output
  k_conv<130, 6, 2><<<dim3(32, NSAMP), 256, 0, stream>>>(
      wp_tail, tail_b, assign, u2, nullptr, nullptr, out, 3, 16, KPAD, 1, 0,
      1, 0);
}
